// MTSQLayer_22136261444314
// MI455X (gfx1250) — matrix operand order repaired and checked by reading, not compiled, not run on hardware
//
#include <hip/hip_runtime.h>

typedef __attribute__((ext_vector_type(16))) _Float16 v16h;
typedef __attribute__((ext_vector_type(8)))  _Float16 v8h;
typedef __attribute__((ext_vector_type(4)))  _Float16 v4h;
typedef __attribute__((ext_vector_type(8)))  float    v8f;
typedef __attribute__((ext_vector_type(4)))  float    v4f;

#define QMIN 0.0f
#define QMAX 255.0f

#define BM 256           // block tile M
#define BN 128           // block tile N
#define BK 32            // K step
#define LDPAD 8
#define LDROW (BK + LDPAD)   // 40 halfs (80B) per LDS row; rows 16B-aligned

// ---------------------------------------------------------------------------
// Kernel 1: fake-quantize + dequantize the weight once, emit f16 copy [N,K].
// q = clamp(round(w/s + zp), 0, 255); dq = s*(q - zp). scale is power-of-two.
// ---------------------------------------------------------------------------
__global__ __launch_bounds__(256) void mtsq_dequant_w_f16(
    const float* __restrict__ w, const float* __restrict__ scale,
    const float* __restrict__ zp, _Float16* __restrict__ wq, int K) {
  int idx = (blockIdx.x * 256 + threadIdx.x) * 4;   // 4 elems, same row (K%4==0)
  int n = idx / K;
  float s  = scale[n];
  float z  = zp[n];
  float rs = 1.0f / s;                               // exact: s is 2^e
  v4f wv = *(const v4f*)(w + idx);
  v4h dq;
#pragma unroll
  for (int i = 0; i < 4; ++i) {
    float q = rintf(wv[i] * rs + z);
    q = fminf(fmaxf(q, QMIN), QMAX);
    dq[i] = (_Float16)(s * (q - z));
  }
  *(v4h*)(wq + idx) = dq;
}

// ---------------------------------------------------------------------------
// Async 16B copy global -> LDS (CDNA5 ASYNCcnt path), via inline asm so it
// works on both toolchains. vdst = LDS byte address, vaddr = 64-bit global.
// ---------------------------------------------------------------------------
__device__ __forceinline__ void async_copy_b128(const void* gptr, void* ldsptr) {
  unsigned lds_off = (unsigned)(uintptr_t)ldsptr;  // low 32 bits = LDS offset
  asm volatile("global_load_async_to_lds_b128 %0, %1, off"
               :: "v"(lds_off), "v"(gptr)
               : "memory");
}
__device__ __forceinline__ void wait_asynccnt0() {
  asm volatile("s_wait_asynccnt 0" ::: "memory");
}

// ---------------------------------------------------------------------------
// Kernel 2: WMMA GEMM  out[M,N] = x[M,K] * wq[N,K]^T + bias[N]
// 256 threads = 8 wave32s. Block tile 256x128, K-step 32, double-buffered LDS.
// Wave grid 4(M) x 2(N); each wave: 4x4 tiles of 16x16 -> 16 WMMA/K-step.
// B tile staged with global_load_async_to_lds_b128; A tile f32->f16 via VALU.
// ---------------------------------------------------------------------------
__global__ __launch_bounds__(256) void mtsq_wmma_gemm(
    const float* __restrict__ x, const _Float16* __restrict__ wq,
    const float* __restrict__ bias, float* __restrict__ out,
    int M, int N, int K) {
  __shared__ _Float16 sA[2][BM * LDROW];   // 2 x 20KB: x tile f16 [row][k]
  __shared__ _Float16 sB[2][BN * LDROW];   // 2 x 10KB: weight tile f16 [n][k]

  const int tid  = threadIdx.x;
  const int lane = tid & 31;
  const int wave = tid >> 5;
  const int wm   = wave >> 1;           // 4 waves along M (64 rows each)
  const int wn   = wave & 1;            // 2 waves along N (64 cols each)

  const int bn = blockIdx.x * BN;       // N outer-fast: 16 blocks share x tile
  const int bm = blockIdx.y * BM;

  v8f acc[4][4];
#pragma unroll
  for (int mi = 0; mi < 4; ++mi)
#pragma unroll
    for (int ni = 0; ni < 4; ++ni)
      acc[mi][ni] = (v8f){};

  const int kTiles = K / BK;

  // ---- staging helpers (inlined by unrolled loops) --------------------------
  auto stage_B_async = [&](int buf, int k0) {
    // 128 x 32 halfs = 512 x 16B chunks, 2 per thread, pure async byte copy
#pragma unroll
    for (int i = 0; i < 2; ++i) {
      const int t   = tid + i * 256;
      const int row = t >> 2;
      const int c   = (t & 3) * 8;
      async_copy_b128(wq + (size_t)(bn + row) * K + k0 + c,
                      &sB[buf][row * LDROW + c]);
    }
  };
  auto stage_A = [&](int buf, int k0) {
    // 256 x 32 floats = 2048 float4 loads, 8 per thread, f32 -> f16 convert
#pragma unroll
    for (int i = 0; i < 8; ++i) {
      const int t   = tid + i * 256;
      const int row = t >> 3;
      const int c   = (t & 7) * 4;
      v4f v = *(const v4f*)(x + (size_t)(bm + row) * K + k0 + c);
      v4h h;
#pragma unroll
      for (int j = 0; j < 4; ++j) h[j] = (_Float16)v[j];
      *(v4h*)(&sA[buf][row * LDROW + c]) = h;
    }
  };

  // ---- prologue: stage tile 0 ----------------------------------------------
  stage_B_async(0, 0);
  stage_A(0, 0);
  wait_asynccnt0();
  __syncthreads();

  for (int kt = 0; kt < kTiles; ++kt) {
    const int cur = kt & 1;
    const int k0  = kt * BK;

    // ---- issue next tile into the other buffer (overlaps with WMMAs) -------
    if (kt + 1 < kTiles) {
      stage_B_async(cur ^ 1, k0 + BK);
      stage_A(cur ^ 1, k0 + BK);
    }
    // light L2 prefetch two tiles ahead for the streamed x operand
    if (kt + 2 < kTiles) {
      __builtin_prefetch(&x[(size_t)(bm + tid) * K + k0 + 2 * BK], 0, 1);
    }

    // ---- B fragments: 32x16, lane = N%16; lanes 0-15 K 0..15, 16-31 K 16..31
    v16h bfrag[4];
#pragma unroll
    for (int ni = 0; ni < 4; ++ni) {
      const int nrow  = wn * 64 + ni * 16 + (lane & 15);
      const int kbase = (lane >> 4) * 16;
      const _Float16* pb = &sB[cur][nrow * LDROW + kbase];
      v8h blo = *(const v8h*)pb;
      v8h bhi = *(const v8h*)(pb + 8);
#pragma unroll
      for (int j = 0; j < 8; ++j) { bfrag[ni][j] = blo[j]; bfrag[ni][8 + j] = bhi[j]; }
    }

    // ---- A fragments streamed + 16 WMMAs. A 16x32: M = lane%16;
    //      lanes 0-15: K 0..15; lanes 16-31: K 16..31.
#pragma unroll
    for (int mi = 0; mi < 4; ++mi) {
      const int mrow  = wm * 64 + mi * 16 + (lane & 15);
      const int kbase = (lane >> 4) * 16;
      const _Float16* pa = &sA[cur][mrow * LDROW + kbase];
      v8h alo = *(const v8h*)pa;
      v8h ahi = *(const v8h*)(pa + 8);
      v16h afrag;
#pragma unroll
      for (int j = 0; j < 8; ++j) { afrag[j] = alo[j]; afrag[8 + j] = ahi[j]; }
#pragma unroll
      for (int ni = 0; ni < 4; ++ni) {
        acc[mi][ni] = __builtin_amdgcn_wmma_f32_16x16x32_f16(
            false, afrag, false, bfrag[ni], (short)0, acc[mi][ni], false, false);
      }
    }

    // async B for next tile must land before anyone reads it after the barrier
    wait_asynccnt0();
    __syncthreads();
  }

  // ---- epilogue: C/D layout — lane = N%16, VGPR v holds M = v + 8*(lane>=16)
#pragma unroll
  for (int ni = 0; ni < 4; ++ni) {
    const int ng = bn + wn * 64 + ni * 16 + (lane & 15);
    const float bv = bias[ng];
#pragma unroll
    for (int mi = 0; mi < 4; ++mi) {
      const int mbase = bm + wm * 64 + mi * 16 + ((lane >> 4) << 3);
#pragma unroll
      for (int v = 0; v < 8; ++v) {
        out[(size_t)(mbase + v) * N + ng] = acc[mi][ni][v] + bv;
      }
    }
  }
}

// ---------------------------------------------------------------------------
extern "C" void kernel_launch(void* const* d_in, const int* in_sizes, int n_in,
                              void* d_out, int out_size, void* d_ws, size_t ws_size,
                              hipStream_t stream) {
  const float* x     = (const float*)d_in[0];
  const float* w     = (const float*)d_in[1];
  const float* bias  = (const float*)d_in[2];
  const float* scale = (const float*)d_in[3];
  const float* zp    = (const float*)d_in[4];
  float* out = (float*)d_out;

  const int DOUT = in_sizes[2];             // 2048
  const int DIN  = in_sizes[1] / DOUT;      // 2048
  const int M    = in_sizes[0] / DIN;       // 16384

  _Float16* wq = (_Float16*)d_ws;           // needs DOUT*DIN*2 = 8 MB

  // 1) fake-quant + dequant weight -> f16 workspace
  {
    const int total = DOUT * DIN;
    dim3 grid(total / (256 * 4));
    mtsq_dequant_w_f16<<<grid, 256, 0, stream>>>(w, scale, zp, wq, DIN);
  }
  // 2) WMMA GEMM with bias (N-dim fastest for L2 reuse of the x tile)
  {
    dim3 grid(DOUT / BN, M / BM);
    mtsq_wmma_gemm<<<grid, 256, 0, stream>>>(x, wq, bias, out, M, DOUT, DIN);
  }
}
